// Router_56925496541861
// MI455X (gfx1250) — matrix operand order repaired and checked by reading, not compiled, not run on hardware
//
#include <hip/hip_runtime.h>
#include <hip/hip_bf16.h>
#include <math.h>

// CDNA5 / gfx1250 wave32 WMMA types
typedef __attribute__((ext_vector_type(16))) __bf16 v16bf;
typedef __attribute__((ext_vector_type(8)))  float  v8f;

#define M_TOTAL   16384          // B*S = 4*4096 tokens
#define D_K       2048           // hidden dim (reduction)
#define N_E       64             // experts
#define M_TILE    128            // tokens per block
#define K_CHUNK   64             // K staged per LDS buffer
#define LDX       72             // padded LDS row (bf16 elems): 144B = 36 dwords
#define NOISE_STD 0.1f

#define W_BUF_BYTES (N_E * LDX * 2)   // 9216 B per W LDS buffer

union __align__(16) Smem {
    struct {
        __bf16 X[2][M_TILE][LDX];   // 2 x 128 x 72 bf16 = 36864 B
        __bf16 W[2][N_E][LDX];      // 2 x  64 x 72 bf16 = 18432 B
    } s;                            // total 55296 B
    float logits[M_TILE][65];       // 33280 B (reused after GEMM)
};

__device__ __forceinline__ void wait_asynccnt0() {
#if __has_builtin(__builtin_amdgcn_s_wait_asynccnt)
    __builtin_amdgcn_s_wait_asynccnt(0);
#else
    asm volatile("s_wait_asynccnt 0x0" ::: "memory");
#endif
}

// one-time W fp32 -> bf16 conversion into workspace (64x2048 = 256KB bf16)
__global__ __launch_bounds__(256)
void convert_w_kernel(const float* __restrict__ Wg, __bf16* __restrict__ Wb)
{
    const int idx = blockIdx.x * 256 + threadIdx.x;   // 0..32767 float4's
    float4 v = ((const float4*)Wg)[idx];
    union { __bf16 h[4]; uint2 u; } p;
    p.h[0] = (__bf16)v.x; p.h[1] = (__bf16)v.y;
    p.h[2] = (__bf16)v.z; p.h[3] = (__bf16)v.w;
    ((uint2*)Wb)[idx] = p.u;
}

__global__ __launch_bounds__(256)
void router_kernel(const float* __restrict__ x,
                   const float* __restrict__ noise,
                   const __bf16* __restrict__ Wb,
                   float* __restrict__ out)
{
    __shared__ Smem sm;

    const int tid      = threadIdx.x;
    const int lane     = tid & 31;
    const int wave     = tid >> 5;
    const int l16      = lane & 15;
    const int lhalf    = lane >> 4;          // 0: lanes 0-15, 1: lanes 16-31
    const int blockRow = blockIdx.x * M_TILE;
    const int waveRow  = wave * 16;          // 8 waves x 16 rows = 128 rows

    // ---- accumulators: 4 N-tiles of 16x16 f32 (experts t*16 .. t*16+15) ----
    const v8f vzero = {0.f,0.f,0.f,0.f,0.f,0.f,0.f,0.f};
    v8f acc[4];
#pragma unroll
    for (int t = 0; t < 4; ++t) acc[t] = vzero;

    // ---- register staging for x global->LDS pipeline ----
    float4 gx[8];   // 128 rows x 64 cols fp32 = 2048 float4 / 256 thr = 8 each

    auto loadX = [&](int c) {
        const int k0 = c * K_CHUNK;
#pragma unroll
        for (int i = 0; i < 8; ++i) {
            int idx = i * 256 + tid;
            int r  = idx >> 4;          // 0..127
            int c4 = idx & 15;          // 0..15 (float4 within row)
            gx[i] = *(const float4*)(x + (size_t)(blockRow + r) * D_K + k0 + c4 * 4);
        }
    };

    auto storeX = [&](int b) {
#pragma unroll
        for (int i = 0; i < 8; ++i) {
            int idx = i * 256 + tid;
            int r  = idx >> 4;
            int c4 = idx & 15;
            union { __bf16 h[4]; uint2 u; } p;
            p.h[0] = (__bf16)gx[i].x; p.h[1] = (__bf16)gx[i].y;
            p.h[2] = (__bf16)gx[i].z; p.h[3] = (__bf16)gx[i].w;
            *(uint2*)&sm.s.X[b][r][c4 * 4] = p.u;
        }
    };

    // ---- async W staging: per-lane address components are loop-invariant ----
    // packet i (i=0,1): idx = i*256+tid -> expert row r = idx>>3, 16B slot c8 = idx&7
    unsigned wGOff[2];   // 32-bit per-lane global byte offset (GVS vaddr)
    unsigned wLDst[2];   // LDS byte offset of destination in buffer 0
#pragma unroll
    for (int i = 0; i < 2; ++i) {
        int idx = i * 256 + tid;
        int r   = idx >> 3;
        int c8  = idx & 7;
        wGOff[i] = (unsigned)((r * D_K + c8 * 8) * 2);  // bytes into bf16 W
        wLDst[i] = (unsigned)(uintptr_t)
            (__attribute__((address_space(3))) __bf16*)&sm.s.W[0][r][c8 * 8];
    }

    // async bf16 W chunk: global -> LDS directly (ASYNCcnt), GVS addressing:
    // saddr = wave-uniform SGPR base (Wb + k0), vaddr = fixed per-lane offset
    auto issueAsyncW = [&](int c, int b) {
        const unsigned long long sbase =
            (unsigned long long)(uintptr_t)Wb + (unsigned long long)(c * K_CHUNK * 2);
        const unsigned bufAdd = b ? (unsigned)W_BUF_BYTES : 0u;
#pragma unroll
        for (int i = 0; i < 2; ++i) {
            asm volatile("global_load_async_to_lds_b128 %0, %1, %2"
                         :: "v"(wLDst[i] + bufAdd),
                            "v"(wGOff[i]),
                            "s"(sbase)
                         : "memory");
        }
    };

    auto compute = [&](int b) {
        // A fragment (16x32 bf16, ISA 7.12.2): lane holds row = lane%16;
        //   elems 0-7  -> K = e      + 8*lhalf   (contiguous 16B)
        //   elems 8-15 -> K = e-8+16 + 8*lhalf   (contiguous 16B)
        // B fragment (32x16 bf16): lane holds expert = lane%16;
        //   same K mapping as A (two contiguous 16B pieces)
        const __bf16* xrow = sm.s.X[b][waveRow + l16];
#pragma unroll
        for (int s = 0; s < 2; ++s) {           // two K=32 steps per chunk
            const int kb = s * 32;
            union Frag { uint4 q[2]; v16bf v; };
            Frag A;
            A.q[0] = *(const uint4*)(xrow + kb + 8 * lhalf);
            A.q[1] = *(const uint4*)(xrow + kb + 16 + 8 * lhalf);
            Frag Bf[4];
#pragma unroll
            for (int t = 0; t < 4; ++t) {       // issue all B loads first
                const __bf16* bp = sm.s.W[b][t * 16 + l16] + kb + 8 * lhalf;
                Bf[t].q[0] = *(const uint4*)(bp);
                Bf[t].q[1] = *(const uint4*)(bp + 16);
            }
#pragma unroll
            for (int t = 0; t < 4; ++t) {       // then the 4 WMMAs
                acc[t] = __builtin_amdgcn_wmma_f32_16x16x32_bf16(
                    /*neg_a=*/false, A.v, /*neg_b=*/false, Bf[t].v,
                    /*c_mod=*/(short)0, acc[t],
                    /*reuse_a=*/false, /*reuse_b=*/false);
            }
        }
    };

    // ---- pipelined GEMM main loop ----
    const int NC = D_K / K_CHUNK;   // 32 chunks
    loadX(0);
    issueAsyncW(0, 0);
    for (int c = 0; c < NC; ++c) {
        const int b = c & 1;
        storeX(b);              // waits x LOADcnt, cvt + ds_store
        wait_asynccnt0();       // W chunk c landed in LDS buffer b
        __syncthreads();
        if (c + 1 < NC) {
            loadX(c + 1);               // x loads in flight during compute
            issueAsyncW(c + 1, b ^ 1);  // W async in flight during compute
        }
        compute(b);
        __syncthreads();
    }

    // ---- add noise, relayout logits into LDS (stride 65 -> conflict free) ----
    // C/D layout: VGPR j, lanes 0-15 -> M=j, N=lane; lanes 16-31 -> M=j+8, N=lane-16
#pragma unroll
    for (int t = 0; t < 4; ++t) {
#pragma unroll
        for (int j = 0; j < 8; ++j) {
            const int M    = j + lhalf * 8;
            const int grow = blockRow + waveRow + M;
            const int e    = t * 16 + l16;
            float lg = acc[t][j] + NOISE_STD * noise[(size_t)grow * N_E + e];
            sm.logits[waveRow + M][e] = lg;
        }
    }
    __syncthreads();

    // ---- per-row softmax + top-2 + scatter (thread-per-row) ----
    if (tid < M_TILE) {
        float*    Lr  = sm.logits[tid];
        const int row = blockRow + tid;

        float l1 = -INFINITY, l2 = -INFINITY;
        int   i1 = 0, i2 = 0;
        for (int e = 0; e < N_E; ++e) {
            float v = Lr[e];
            if (v > l1)      { l2 = l1; i2 = i1; l1 = v; i1 = e; }
            else if (v > l2) { l2 = v; i2 = e; }
        }
        const float m = l1;                   // row max

        float sum = 0.f;
        for (int e = 0; e < N_E; ++e) {
            float ex = __expf(Lr[e] - m);
            Lr[e] = ex;                       // cache exp for probs pass
            sum += ex;
        }
        const float inv  = 1.f / sum;
        const float p1   = inv;               // exp(l1-m)=1
        const float p2   = __expf(l2 - m) * inv;
        const float wsum = p1 + p2;
        const float w1   = p1 / wsum;
        const float w2   = p2 / wsum;

        float* dispatch = out;
        float* probs    = out + (size_t)M_TOTAL * N_E;
        float* wout     = out + (size_t)2 * M_TOTAL * N_E;
        float* sout     = wout + (size_t)M_TOTAL * 2;

        for (int e = 0; e < N_E; ++e) {
            probs[(size_t)row * N_E + e] = Lr[e] * inv;
            float d = (e == i1) ? w1 : ((e == i2) ? w2 : 0.f);
            dispatch[(size_t)row * N_E + e] = d;
        }
        wout[(size_t)row * 2 + 0] = w1;
        wout[(size_t)row * 2 + 1] = w2;
        sout[(size_t)row * 2 + 0] = (float)i1;
        sout[(size_t)row * 2 + 1] = (float)i2;
    }
}

extern "C" void kernel_launch(void* const* d_in, const int* in_sizes, int n_in,
                              void* d_out, int out_size, void* d_ws, size_t ws_size,
                              hipStream_t stream) {
    const float* x     = (const float*)d_in[0];
    const float* noise = (const float*)d_in[1];
    const float* W     = (const float*)d_in[2];
    float* out = (float*)d_out;
    __bf16* Wb = (__bf16*)d_ws;                 // 64*2048*2 = 256KB scratch

    convert_w_kernel<<<dim3(128), dim3(256), 0, stream>>>(W, Wb);
    router_kernel<<<dim3(M_TOTAL / M_TILE), dim3(256), 0, stream>>>(x, noise, Wb, out);
}
